// MinEncoderOrder_57990648430804
// MI455X (gfx1250) — matrix operand order repaired and checked by reading, not compiled, not run on hardware
//
#include <hip/hip_runtime.h>

typedef _Float16 v16h __attribute__((ext_vector_type(16)));
typedef _Float16 v8h  __attribute__((ext_vector_type(8)));
typedef float    v8f  __attribute__((ext_vector_type(8)));

#define TILE_M 256   // 8 waves x 32 rows
#define TILE_N 32    // two 16-wide WMMA sub-tiles
#define TILE_K 32

__device__ __forceinline__ float mish_pos(float r) {
  // r >= 0 (post-relu). mish(r) = r * tanh(softplus(r)); softplus stable for large r.
  float sp = (r > 15.f) ? r : log1pf(__expf(r));
  return r * tanhf(sp);
}

// Per-column epilogue for one 16x16 accumulator tile.
__device__ __forceinline__ void store_tile(v8f c, int ng, int N, int rbase,
    const float* __restrict__ bias,
    _Float16* __restrict__ Dout, int ldo, int doff, int ocols, int mode,
    _Float16* __restrict__ Dcar, int ldc, float& sum, float& ssq) {
  if (ng >= N) return;
  const float bv = bias[ng];
  #pragma unroll
  for (int v = 0; v < 8; ++v) {
    const int rg = rbase + v;
    float val = c[v] + bv;
    if (mode == 0) {
      float r = fmaxf(val, 0.f);
      if (ng < ocols) {
        float y = mish_pos(r);
        Dout[(size_t)rg * ldo + doff + ng] = (_Float16)y;
        sum += y; ssq += y * y;
      } else {
        Dcar[(size_t)rg * ldc + (ng - ocols)] = (_Float16)r;
      }
    } else {
      Dout[(size_t)rg * ldo + doff + ng] = (_Float16)val;
    }
  }
}

// ---------------------------------------------------------------------------
// Weight prep: W (KxN f32, row-major) -> Wt (Npad x Kpad f16, transposed,
// zero-padded, scaled by bnp.scale); bias2 = bias + bnp.shift * colsum(W).
// Folding BN of the *input* activation into W/bias:
//   (a*s + t) @ W = a @ (s*W) + t * colsum(W)
// One block per output row n of Wt (64 threads).
// ---------------------------------------------------------------------------
__global__ void prep_w_kernel(const float* __restrict__ W,
                              const float* __restrict__ bias,
                              const float* __restrict__ bnp,
                              int K, int N, int Kpad,
                              _Float16* __restrict__ Wt,
                              float* __restrict__ bias2) {
  const int n = blockIdx.x;        // 0..Npad-1
  const int t = threadIdx.x;       // 64 threads
  float scale = 1.f, shift = 0.f;
  if (bnp) { scale = bnp[0]; shift = bnp[1]; }
  float csum = 0.f;
  for (int k = t; k < Kpad; k += 64) {
    float w = 0.f;
    if (k < K && n < N) { w = W[(size_t)k * N + n]; csum += w; }
    Wt[(size_t)n * Kpad + k] = (_Float16)(w * scale);
  }
  __shared__ float sr[64];
  sr[t] = csum;
  __syncthreads();
  #pragma unroll
  for (int s = 32; s > 0; s >>= 1) {
    if (t < s) sr[t] += sr[t + s];
    __syncthreads();
  }
  if (t == 0 && n < N) bias2[n] = bias[n] + shift * sr[0];
}

// Pad/convert x (B x 24 f32) into xh (B x 32 f16).
__global__ void xconv_kernel(const float* __restrict__ x, _Float16* __restrict__ xh,
                             int total /* B*32 */) {
  int i = blockIdx.x * 256 + threadIdx.x;
  if (i >= total) return;
  int r = i >> 5, c = i & 31;
  xh[i] = (c < 24) ? (_Float16)x[(size_t)r * 24 + c] : (_Float16)0.f;
}

// ---------------------------------------------------------------------------
// WMMA GEMM, f16 in / f32 accum, "split" epilogue.
//   D = A(row-major f16, lda) @ Wt^T + bias2        (Wt is [Npad][Kpad] f16)
//   mode 0: relu; cols [0,ocols) -> mish -> Dout at col doff (+stats)
//           cols [ocols,N)       -> raw relu carry -> Dcar at col (n-ocols)
//   mode 1: linear -> Dout
// Kpad multiple of 32; Wt zero-padded in K and N => no bounds checks in loop
// (A's K-padding garbage is multiplied by W's zero rows).
// Block = 256 thr = 8 waves; block tile 256(M) x 32(N); wave tile 32x32:
// 2 A-frags (LDS) x 2 B-frags (direct L2-resident global) -> 4 WMMA / step.
// ---------------------------------------------------------------------------
__global__ __launch_bounds__(256)
void wmma_gemm(const _Float16* __restrict__ A, int lda,
               const _Float16* __restrict__ Wt, int ldw,
               const float* __restrict__ bias2, int Kpad, int N,
               _Float16* __restrict__ Dout, int ldo, int doff, int ocols, int mode,
               _Float16* __restrict__ Dcar, int ldc,
               float* __restrict__ stats) {
  __shared__ _Float16 sA[TILE_M][TILE_K + 8];   // 80B row stride: conflict-free frags
  __shared__ float sred[512];

  const int t    = threadIdx.x;
  const int n0   = blockIdx.x * TILE_N;
  const int row0 = blockIdx.y * TILE_M;

  const int wv   = t >> 5;
  const int lane = t & 31;
  const int mrow = lane & 15;           // A frag M within 16
  const int kbA  = (lane >> 4) << 3;    // A frag K base: 0 or 8
  const int nloc = lane & 15;           // B/C/D column within 16-wide sub-tile
  const int kkB  = (lane >> 4) << 3;    // B frag K base: 0 or 8

  const _Float16* wp0 = Wt + (size_t)(n0 + nloc) * ldw + kkB;
  const _Float16* wp1 = Wt + (size_t)(n0 + 16 + nloc) * ldw + kkB;

  v8f c00 = {}, c01 = {}, c10 = {}, c11 = {};

  for (int k0 = 0; k0 < Kpad; k0 += TILE_K) {
    // A tile: 256 rows x 32 halves -> 4 x v8h per thread, fully vectorized.
    #pragma unroll
    for (int j = 0; j < 4; ++j) {
      int i = t + 256 * j;              // 0..1023
      int r = i >> 2;                   // row 0..255
      int seg = i & 3;                  // 8-half segment
      *(v8h*)&sA[r][seg * 8] =
          *(const v8h*)&A[(size_t)(row0 + r) * lda + k0 + seg * 8];
    }
    __syncthreads();

    // B fragments straight from L2-resident padded Wt (16B aligned rows).
    v8h b0_lo = *(const v8h*)(wp0 + k0);
    v8h b0_hi = *(const v8h*)(wp0 + k0 + 16);
    v8h b1_lo = *(const v8h*)(wp1 + k0);
    v8h b1_hi = *(const v8h*)(wp1 + k0 + 16);

    // Two A fragments from LDS.
    const _Float16* ar0 = &sA[wv * 32 + mrow][0];
    const _Float16* ar1 = &sA[wv * 32 + 16 + mrow][0];
    v8h a0_lo = *(const v8h*)(ar0 + kbA);
    v8h a0_hi = *(const v8h*)(ar0 + 16 + kbA);
    v8h a1_lo = *(const v8h*)(ar1 + kbA);
    v8h a1_hi = *(const v8h*)(ar1 + 16 + kbA);

    v16h af0, af1, bf0, bf1;
    #pragma unroll
    for (int h = 0; h < 8; ++h) {
      af0[h] = a0_lo[h]; af0[8 + h] = a0_hi[h];
      af1[h] = a1_lo[h]; af1[8 + h] = a1_hi[h];
      bf0[h] = b0_lo[h]; bf0[8 + h] = b0_hi[h];
      bf1[h] = b1_lo[h]; bf1[8 + h] = b1_hi[h];
    }
    c00 = __builtin_amdgcn_wmma_f32_16x16x32_f16(false, af0, false, bf0, (short)0, c00, false, false);
    c01 = __builtin_amdgcn_wmma_f32_16x16x32_f16(false, af0, false, bf1, (short)0, c01, false, false);
    c10 = __builtin_amdgcn_wmma_f32_16x16x32_f16(false, af1, false, bf0, (short)0, c10, false, false);
    c11 = __builtin_amdgcn_wmma_f32_16x16x32_f16(false, af1, false, bf1, (short)0, c11, false, false);
    __syncthreads();
  }

  // ---------------- epilogue --------------------------------------------------
  float sum = 0.f, ssq = 0.f;
  const int rb0 = row0 + wv * 32 + ((lane >> 4) << 3);   // D: M = v + 8*(lane>=16)
  const int rb1 = rb0 + 16;
  store_tile(c00, n0 + nloc,      N, rb0, bias2, Dout, ldo, doff, ocols, mode, Dcar, ldc, sum, ssq);
  store_tile(c01, n0 + 16 + nloc, N, rb0, bias2, Dout, ldo, doff, ocols, mode, Dcar, ldc, sum, ssq);
  store_tile(c10, n0 + nloc,      N, rb1, bias2, Dout, ldo, doff, ocols, mode, Dcar, ldc, sum, ssq);
  store_tile(c11, n0 + 16 + nloc, N, rb1, bias2, Dout, ldo, doff, ocols, mode, Dcar, ldc, sum, ssq);

  if (stats) {  // uniform per launch
    sred[t] = sum; sred[256 + t] = ssq;
    __syncthreads();
    #pragma unroll
    for (int s = 128; s > 0; s >>= 1) {
      if (t < s) { sred[t] += sred[t + s]; sred[256 + t] += sred[256 + t + s]; }
      __syncthreads();
    }
    if (t == 0) {
      atomicAdd(&stats[0], sred[0]);
      atomicAdd(&stats[1], sred[256]);
    }
  }
}

__global__ void zero_stats_kernel(float* __restrict__ s, int n) {
  int i = threadIdx.x;
  if (i < n) s[i] = 0.f;
}

// Fold whole-tensor BN stats into an affine (scale, shift) for the next prep_w.
__global__ void bn_finalize_kernel(const float* __restrict__ stats,
                                   const float* __restrict__ gamma,
                                   const float* __restrict__ beta,
                                   float inv_count, float* __restrict__ bnp) {
  float mean = stats[0] * inv_count;
  float var  = stats[1] * inv_count - mean * mean;
  float inv  = rsqrtf(var + 1e-5f);
  float sc   = gamma[0] * inv;
  bnp[0] = sc;
  bnp[1] = beta[0] - mean * sc;
}

// sigmoid + per-feature 8-entry codebook argmin + 3-bit unpack to float.
__global__ __launch_bounds__(256)
void quantize_kernel(const _Float16* __restrict__ z, const float* __restrict__ cb,
                     float* __restrict__ out, int total) {
  int i = blockIdx.x * 256 + threadIdx.x;   // i = b*512 + j
  if (i >= total) return;
  int j = i & 511;
  float s = 1.f / (1.f + __expf(-(float)z[i]));
  const float* c = cb + j * 8;
  int best = 0; float bd = 3.4e38f;
  #pragma unroll
  for (int k = 0; k < 8; ++k) {
    float d = s - c[k]; d = d * d;
    if (d < bd) { bd = d; best = k; }   // strict < keeps first min (argmin semantics)
  }
  float* o = out + (size_t)i * 3;
  o[0] = (float)((best >> 2) & 1);
  o[1] = (float)((best >> 1) & 1);
  o[2] = (float)(best & 1);
}

// ---------------------------------------------------------------------------
// Inputs (flattened setup_inputs order):
//  0: x(B,24) | 1..9: split W | 10..18: split b | 19..21: gamma | 22..24: beta
//  25: fc_w(600,512) | 26: fc_b(512) | 27: codebook(512*8)
// ---------------------------------------------------------------------------
extern "C" void kernel_launch(void* const* d_in, const int* in_sizes, int n_in,
                              void* d_out, int out_size, void* d_ws, size_t ws_size,
                              hipStream_t stream) {
  (void)n_in; (void)out_size; (void)ws_size;
  const float* x = (const float*)d_in[0];
  const float* Wm[9]; const float* bm[9];
  for (int i = 0; i < 9; ++i) { Wm[i] = (const float*)d_in[1 + i]; bm[i] = (const float*)d_in[10 + i]; }
  const float* gamma[3]; const float* beta[3];
  for (int i = 0; i < 3; ++i) { gamma[i] = (const float*)d_in[19 + i]; beta[i] = (const float*)d_in[22 + i]; }
  const float* fcw = (const float*)d_in[25];
  const float* fcb = (const float*)d_in[26];
  const float* cb  = (const float*)d_in[27];

  const int Brows = in_sizes[0] / 24;   // 65536

  // ws layout: stats(6f) | bnp(6f) | f16: xh(B,32) t1(B,608) t2(B,304) o0(B,608)
  //            o1(B,608) Wt(608*608) | bias2(608f)
  char* base = (char*)d_ws;
  float* stats = (float*)base;
  float* bnp   = (float*)(base + 256);
  _Float16* xh = (_Float16*)(base + 512);
  _Float16* t1 = xh + (size_t)Brows * 32;
  _Float16* t2 = t1 + (size_t)Brows * 608;
  _Float16* o0 = t2 + (size_t)Brows * 304;
  _Float16* o1 = o0 + (size_t)Brows * 608;
  _Float16* Wt = o1 + (size_t)Brows * 608;
  float* bias2 = (float*)(Wt + (size_t)608 * 608);

  zero_stats_kernel<<<1, 32, 0, stream>>>(stats, 6);
  xconv_kernel<<<(Brows * 32 + 255) / 256, 256, 0, stream>>>(x, xh, Brows * 32);

  auto kp = [](int K) { return (K + 31) & ~31; };
  auto np = [](int N) { return (N + 31) & ~31; };
  dim3 blk(256);

  // One GEMM stage = prep_w (transpose/pad/scale-fold) + WMMA GEMM.
  auto stage = [&](const _Float16* A, int lda, const float* W, const float* b,
                   const float* bn, int K, int N,
                   _Float16* Dout, int ldo, int doff, int ocols, int mode,
                   _Float16* Dcar, int ldc, float* st) {
    const int Kp = kp(K), Np = np(N);
    prep_w_kernel<<<Np, 64, 0, stream>>>(W, b, bn, K, N, Kp, Wt, bias2);
    dim3 g((unsigned)(Np / TILE_N), (unsigned)(Brows / TILE_M));
    wmma_gemm<<<g, blk, 0, stream>>>(A, lda, Wt, Kp, bias2, Kp, N,
                                     Dout, ldo, doff, ocols, mode, Dcar, ldc, st);
  };

  // ---- Layer 1: (24->300 |150), (150->150 |100), (50->50 |50) ----
  stage(xh, 32, Wm[0], bm[0], nullptr, 24, 300, o0, 608, 0, 150, 0, t1, 608, stats + 0);
  stage(t1, 608, Wm[1], bm[1], nullptr, 150, 150, o0, 608, 150, 100, 0, t2, 304, stats + 0);
  stage(t2, 304, Wm[2], bm[2], nullptr, 50, 50, o0, 608, 250, 50, 0, t2, 304, stats + 0);
  bn_finalize_kernel<<<1, 1, 0, stream>>>(stats + 0, gamma[0], beta[0],
      1.f / ((float)Brows * 300.f), bnp + 0);

  // ---- Layer 2 (BN0 folded into first-stage weights) ----
  stage(o0, 608, Wm[3], bm[3], bnp + 0, 300, 600, o1, 608, 0, 300, 0, t1, 608, stats + 2);
  stage(t1, 608, Wm[4], bm[4], nullptr, 300, 300, o1, 608, 300, 200, 0, t2, 304, stats + 2);
  stage(t2, 304, Wm[5], bm[5], nullptr, 100, 100, o1, 608, 500, 100, 0, t2, 304, stats + 2);
  bn_finalize_kernel<<<1, 1, 0, stream>>>(stats + 2, gamma[1], beta[1],
      1.f / ((float)Brows * 600.f), bnp + 2);

  // ---- Layer 3 (BN1 folded; writes back into o0) ----
  stage(o1, 608, Wm[6], bm[6], bnp + 2, 600, 600, o0, 608, 0, 300, 0, t1, 608, stats + 4);
  stage(t1, 608, Wm[7], bm[7], nullptr, 300, 300, o0, 608, 300, 200, 0, t2, 304, stats + 4);
  stage(t2, 304, Wm[8], bm[8], nullptr, 100, 100, o0, 608, 500, 100, 0, t2, 304, stats + 4);
  bn_finalize_kernel<<<1, 1, 0, stream>>>(stats + 4, gamma[2], beta[2],
      1.f / ((float)Brows * 600.f), bnp + 4);

  // ---- FC: (600->512) linear, BN2 folded; z -> t1 (f16, ld 512) ----
  stage(o0, 608, fcw, fcb, bnp + 4, 600, 512, t1, 512, 0, 512, 1, t2, 304, nullptr);

  // ---- sigmoid + codebook argmin + bit unpack ----
  const int total = Brows * 512;
  quantize_kernel<<<(total + 255) / 256, 256, 0, stream>>>(t1, cb, (float*)d_out, total);
}
